// ConvLRUBlock_49392123904702
// MI455X (gfx1250) — compile-verified, matrix operand order checked by reading, not run on hardware
//
#include <hip/hip_runtime.h>
#include <hip/hip_bf16.h>

// ---------------------------------------------------------------------------
// ConvLRU block for MI455X (gfx1250, wave32, WMMA).
//  B=1, C=64, L=32, H=128, W=128, K=7, PAD=3.
// Pipeline: prep(weights->bf16, fuse depth conv) -> implicit-GEMM WMMA conv
//           (async-to-LDS double-buffered weight staging) ->
//           RMSNorm+GLU (WMMA) -> LRU scan -> w_out GEMM + residual (WMMA).
// d_out is reused as the intermediate f32 buffer (only ~0.5MB of d_ws used).
// ---------------------------------------------------------------------------

typedef __bf16 bf16;
typedef bf16  v16bf __attribute__((ext_vector_type(16)));
typedef float v8f   __attribute__((ext_vector_type(8)));
typedef int   v4i   __attribute__((ext_vector_type(4)));

#define Cc   64
#define Ll   32
#define Hh   128
#define Ww   128
#define KK   7
#define TAPS 49
#define CS   (Ll * Hh * Ww)   // channel stride   = 524288
#define LS   (Hh * Ww)        // time step stride = 16384
#define PIXELS (Ll * Hh * Ww) // 524288

// gfx1250 async global->LDS copy path (ASYNCcnt), guarded per-toolchain.
#if defined(__has_builtin)
#  if __has_builtin(__builtin_amdgcn_global_load_async_to_lds_b128) && \
      __has_builtin(__builtin_amdgcn_s_wait_asynccnt)
#    define HAVE_ASYNC_LDS 1
#  endif
#endif
#ifndef HAVE_ASYNC_LDS
#  define HAVE_ASYNC_LDS 0
#endif

#define AS1 __attribute__((address_space(1)))
#define AS3 __attribute__((address_space(3)))

#if HAVE_ASYNC_LDS
// Per-lane 16-byte async copy: global -> LDS, tracked by ASYNCcnt.
// Builtin signature (from hipcc diagnostic): (AS1 v4i*, AS3 v4i*, imm, imm).
static __device__ inline void async_copy16(const bf16* g, bf16* l) {
  __builtin_amdgcn_global_load_async_to_lds_b128(
      (AS1 v4i*)(v4i*)g, (AS3 v4i*)(v4i*)l, 0, 0);
}
#endif

static __device__ inline v8f zero8() {
  v8f z;
#pragma unroll
  for (int i = 0; i < 8; ++i) z[i] = 0.0f;
  return z;
}

// Load a 16-bit A/B fragment (16x32 MxK or 32x16 KxN, row-of-K per lane).
// `row` points at this lane's K=0 element; K pairs are contiguous -> b32 loads.
// CDNA5 layout: elem e (v=e>>1,pos=e&1): K = (v>=4?16:0) + 8*half + 2*(v&3)+pos
static __device__ inline v16bf load_frag16(const bf16* row, int half) {
  v16bf f;
#pragma unroll
  for (int v = 0; v < 8; ++v) {
    const int k0 = ((v & 4) ? 16 : 0) + half * 8 + 2 * (v & 3);
    union { unsigned u; bf16 h[2]; } cv;
    cv.u = *(const unsigned*)(row + k0);
    f[2 * v]     = cv.h[0];
    f[2 * v + 1] = cv.h[1];
  }
  return f;
}

static __device__ inline v8f wmma_bf16(v16bf a, v16bf b, v8f c) {
  return __builtin_amdgcn_wmma_f32_16x16x32_bf16(
      /*neg_a=*/false, a, /*neg_b=*/false, b,
      /*c_mod=*/(short)0, c, /*reuse_a=*/false, /*reuse_b=*/false);
}

// ---------------------------------------------------------------------------
// Kernel 0: weight prep.
//  w_eff[t][o][i] = sum_m w_depth[o][m] * w_spatial[m][i][ky][kx]   (bf16)
//  plus bf16 conversion of w_in (128x64) and w_out (64x64).
// ---------------------------------------------------------------------------
__global__ __launch_bounds__(256) void prep_weights(
    const float* __restrict__ w_spatial,  // [C][C][7][7]
    const float* __restrict__ w_depth,    // [C][C]
    const float* __restrict__ w_in,       // [2C][C]
    const float* __restrict__ w_out,      // [C][C]
    bf16* __restrict__ weff,              // [49][64][64]
    bf16* __restrict__ win_bf,            // [128][64]
    bf16* __restrict__ wout_bf)           // [64][64]
{
  const int gid = blockIdx.x * 256 + threadIdx.x;
  if (gid < TAPS * Cc * Cc) {
    const int t   = gid >> 12;          // /4096
    const int rem = gid & 4095;
    const int o   = rem >> 6;
    const int i   = rem & 63;
    float s = 0.0f;
#pragma unroll 8
    for (int m = 0; m < Cc; ++m)
      s += w_depth[o * Cc + m] * w_spatial[(m * Cc + i) * TAPS + t];
    weff[gid] = (bf16)s;
  } else if (gid < TAPS * Cc * Cc + 2 * Cc * Cc) {
    const int j = gid - TAPS * Cc * Cc;
    win_bf[j] = (bf16)w_in[j];
  } else if (gid < TAPS * Cc * Cc + 3 * Cc * Cc) {
    const int j = gid - (TAPS * Cc * Cc + 2 * Cc * Cc);
    wout_bf[j] = (bf16)w_out[j];
  }
}

// ---------------------------------------------------------------------------
// Kernel 1: fused 7x7 spatial conv + depth conv (weights pre-fused) + b_depth.
// One WG per (l, h, 32-wide w strip). 8 waves = 4 c_out tiles x 2 pixel tiles.
// Implicit GEMM: M=c_out(16), N=pixels(16), K=c_in*49 in 32-chunks, bf16 WMMA.
// Per-tap 8KB weight chunk is double-buffered in LDS and fetched with
// GLOBAL_LOAD_ASYNC_TO_LDS_B128 (overlapped with the current tap's WMMAs).
// ---------------------------------------------------------------------------
#define PPITCH 66   // 64 channels + 2 pad (bank-conflict avoidance)
__global__ __launch_bounds__(256) void conv_spatial(
    const float* __restrict__ x,       // [C][L][H][W]
    const bf16* __restrict__ weff,     // [49][64][64]
    const float* __restrict__ b_depth, // [C]
    float* __restrict__ out)           // [C][L][H][W]
{
  __shared__ bf16 patch[7 * 38 * PPITCH];  // [hh][ww][c] (c fastest)
  __shared__ bf16 Ash[2][Cc * Cc];         // double-buffered tap weights

  const int bid  = blockIdx.x;
  const int w0   = (bid & 3) * 32;
  const int h    = (bid >> 2) & (Hh - 1);
  const int l    = bid >> 9;
  const int tid  = threadIdx.x;
  const int lane = tid & 31;
  const int wave = tid >> 5;
  const int co0  = (wave & 3) * 16;   // c_out tile base
  const int p0   = (wave >> 2) * 16;  // pixel tile base (within 32 strip)
  const int half = lane >> 4;
  const int lm   = lane & 15;

#if HAVE_ASYNC_LDS
  // Kick off tap 0's weight chunk before touching the input patch.
  async_copy16(weff + tid * 8, &Ash[0][tid * 8]);
  async_copy16(weff + 2048 + tid * 8, &Ash[0][2048 + tid * 8]);
#endif

  // Stage padded input patch: H edge-clamped, W wrapped; f32 -> bf16.
  for (int idx = tid; idx < 7 * 38 * Cc; idx += 256) {
    const int c   = idx / (7 * 38);
    const int rem = idx - c * (7 * 38);
    const int hh  = rem / 38;
    const int ww  = rem - hh * 38;
    const int gh  = min(max(h + hh - 3, 0), Hh - 1);
    const int gw  = (w0 + ww - 3 + Ww) & (Ww - 1);
    patch[(hh * 38 + ww) * PPITCH + c] =
        (bf16)x[c * CS + l * LS + gh * Ww + gw];
  }

#if !HAVE_ASYNC_LDS
  // Fallback: synchronous staging of tap 0.
  {
    const unsigned* src = (const unsigned*)weff;
    for (int i = tid; i < (Cc * Cc) / 2; i += 256)
      ((unsigned*)Ash[0])[i] = src[i];
  }
#else
  __builtin_amdgcn_s_wait_asynccnt(0);
#endif
  __syncthreads();

  v8f acc = zero8();

  for (int t = 0; t < TAPS; ++t) {
    const int cur = t & 1;
#if HAVE_ASYNC_LDS
    // Overlap: fetch next tap's chunk into the other buffer (ASYNCcnt).
    if (t + 1 < TAPS) {
      const bf16* nsrc = weff + (t + 1) * (Cc * Cc);
      async_copy16(nsrc + tid * 8, &Ash[cur ^ 1][tid * 8]);
      async_copy16(nsrc + 2048 + tid * 8, &Ash[cur ^ 1][2048 + tid * 8]);
    }
#endif

    const int ky = t / KK;
    const int kx = t - ky * KK;
#pragma unroll
    for (int s = 0; s < 2; ++s) {
      const int ci0 = s * 32;
      v16bf a = load_frag16(&Ash[cur][(co0 + lm) * Cc + ci0], half);
      v16bf b = load_frag16(&patch[(ky * 38 + (p0 + lm + kx)) * PPITCH + ci0],
                            half);
      acc = wmma_bf16(a, b, acc);
    }

    if (t + 1 < TAPS) {
#if HAVE_ASYNC_LDS
      __builtin_amdgcn_s_wait_asynccnt(0);
      __syncthreads();  // next buffer ready for all waves; cur free to reuse
#else
      __syncthreads();  // readers of cur done
      const unsigned* src = (const unsigned*)(weff + (t + 1) * (Cc * Cc));
      for (int i = tid; i < (Cc * Cc) / 2; i += 256)
        ((unsigned*)Ash[cur ^ 1])[i] = src[i];
      if (t + 2 < TAPS)
        __builtin_prefetch(weff + (t + 2) * (Cc * Cc), 0, 0);
      __syncthreads();
#endif
    }
  }

  // Epilogue: + b_depth, store f32 (C/D layout: M = r + 8*half, N = lm).
#pragma unroll
  for (int r = 0; r < 8; ++r) {
    const int co = co0 + r + 8 * half;
    out[co * CS + l * LS + h * Ww + (w0 + p0 + lm)] = acc[r] + b_depth[co];
  }
}

// ---------------------------------------------------------------------------
// Kernel 2: RMSNorm (channel dim) + w_in(128x64) GEMM + b_in + GLU + gamma.
// One WG per 16 pixels; in-place on buf. 8 waves = 8 row-tiles of w_in.
// ---------------------------------------------------------------------------
__global__ __launch_bounds__(256) void pointwise_in_glu(
    float* __restrict__ buf,              // [C][L*H*W], in/out
    const bf16* __restrict__ win_bf,      // [128][64]
    const float* __restrict__ b_in,       // [128]
    const float* __restrict__ rms_weight, // [64]
    const float* __restrict__ nu_log)     // [64]
{
  __shared__ float vraw[16 * Cc];      // [p][c]
  __shared__ float rinv[16];
  __shared__ float gam[Cc];
  __shared__ bf16  vn[16 * Cc];        // normalized, [p][c]
  __shared__ float yt[2 * Cc * 16];    // [row(0..127)][p]

  const int tid  = threadIdx.x;
  const int lane = tid & 31;
  const int wave = tid >> 5;
  const int half = lane >> 4;
  const int lm   = lane & 15;
  const long pix0 = (long)blockIdx.x * 16;

  for (int idx = tid; idx < 16 * Cc; idx += 256) {
    const int c = idx >> 4, p = idx & 15;
    vraw[p * Cc + c] = buf[(long)c * CS + pix0 + p];
  }
  __syncthreads();

  if (tid < 16) {
    float ss = 0.0f;
#pragma unroll 8
    for (int c = 0; c < Cc; ++c) { float v = vraw[tid * Cc + c]; ss += v * v; }
    rinv[tid] = rsqrtf(ss * (1.0f / Cc) + 1e-6f);
  }
  if (tid < Cc) {  // LRU input normalization gamma = sqrt(1 - |lam|^2)
    const float a2 = expf(-2.0f * expf(nu_log[tid]));
    gam[tid] = sqrtf(1.0f - a2);
  }
  __syncthreads();

  for (int idx = tid; idx < 16 * Cc; idx += 256) {
    const int c = idx >> 4, p = idx & 15;
    vn[p * Cc + c] = (bf16)(vraw[p * Cc + c] * rinv[p] * rms_weight[c]);
  }
  __syncthreads();

  // y = w_in @ vnorm + b_in  (each wave: rows 16*wave .. 16*wave+15)
  const int r0 = wave * 16;
  v8f acc = zero8();
#pragma unroll
  for (int s = 0; s < 2; ++s) {
    const int ci0 = s * 32;
    v16bf a = load_frag16(&win_bf[(r0 + lm) * Cc + ci0], half);
    v16bf b = load_frag16(&vn[lm * Cc + ci0], half);
    acc = wmma_bf16(a, b, acc);
  }
#pragma unroll
  for (int r = 0; r < 8; ++r) {
    const int row = r0 + r + 8 * half;
    yt[row * 16 + lm] = acc[r] + b_in[row];
  }
  __syncthreads();

  // GLU + gamma, write back in place.
  for (int idx = tid; idx < 16 * Cc; idx += 256) {
    const int c = idx >> 4, p = idx & 15;
    const float x1 = yt[c * 16 + p];
    const float x2 = yt[(c + Cc) * 16 + p];
    const float g  = x1 * (1.0f / (1.0f + expf(-x2))) * gam[c];
    buf[(long)c * CS + pix0 + p] = g;
  }
}

// ---------------------------------------------------------------------------
// Kernel 3: diagonal complex LRU scan over L (in-place, real part stored).
// One thread per (c,h,w): h_t = lam*h_{t-1} + u_t, lam = exp(-e^nu + i e^th).
// ---------------------------------------------------------------------------
__global__ __launch_bounds__(256) void lru_scan(
    float* __restrict__ buf,              // [C][L][H*W]
    const float* __restrict__ nu_log,
    const float* __restrict__ theta_log)
{
  const int s  = blockIdx.x * 256 + threadIdx.x;   // 0 .. C*H*W-1
  const int c  = s >> 14;
  const int hw = s & (LS - 1);
  const float mod = expf(-expf(nu_log[c]));
  const float th  = expf(theta_log[c]);
  const float lr  = mod * cosf(th);
  const float li  = mod * sinf(th);
  float hr = 0.0f, hi = 0.0f;
  const long base = (long)c * CS + hw;
#pragma unroll 4
  for (int t = 0; t < Ll; ++t) {
    const float u  = buf[base + (long)t * LS];
    const float nr = fmaf(lr, hr, fmaf(-li, hi, u));
    const float ni = fmaf(lr, hi, li * hr);
    hr = nr; hi = ni;
    buf[base + (long)t * LS] = hr;
  }
}

// ---------------------------------------------------------------------------
// Kernel 4: w_out(64x64) GEMM + b_out + residual. One WG per 32 pixels.
// 8 waves = 2 pixel tiles x 4 row tiles. Reads own pixels, then overwrites.
// ---------------------------------------------------------------------------
__global__ __launch_bounds__(256) void pointwise_out_res(
    float* __restrict__ buf,            // in: h.real, out: final (== d_out)
    const float* __restrict__ x,        // residual
    const bf16* __restrict__ wout_bf,   // [64][64]
    const float* __restrict__ b_out)    // [64]
{
  __shared__ bf16 vn[32 * Cc];  // [p][c]

  const int tid  = threadIdx.x;
  const int lane = tid & 31;
  const int wave = tid >> 5;
  const int half = lane >> 4;
  const int lm   = lane & 15;
  const long pix0 = (long)blockIdx.x * 32;

  for (int idx = tid; idx < 32 * Cc; idx += 256) {
    const int c = idx >> 5, p = idx & 31;
    vn[p * Cc + c] = (bf16)buf[(long)c * CS + pix0 + p];
  }
  __syncthreads();

  const int r0 = (wave & 3) * 16;   // c_out tile
  const int p0 = (wave >> 2) * 16;  // pixel tile
  v8f acc = zero8();
#pragma unroll
  for (int s = 0; s < 2; ++s) {
    const int ci0 = s * 32;
    v16bf a = load_frag16(&wout_bf[(r0 + lm) * Cc + ci0], half);
    v16bf b = load_frag16(&vn[(p0 + lm) * Cc + ci0], half);
    acc = wmma_bf16(a, b, acc);
  }
#pragma unroll
  for (int r = 0; r < 8; ++r) {
    const int  co = r0 + r + 8 * half;
    const long q  = pix0 + p0 + lm;
    buf[(long)co * CS + q] = acc[r] + b_out[co] + x[(long)co * CS + q];
  }
}

// ---------------------------------------------------------------------------
extern "C" void kernel_launch(void* const* d_in, const int* in_sizes, int n_in,
                              void* d_out, int out_size, void* d_ws,
                              size_t ws_size, hipStream_t stream) {
  const float* x          = (const float*)d_in[0];
  const float* w_spatial  = (const float*)d_in[1];
  const float* w_depth    = (const float*)d_in[2];
  const float* b_depth    = (const float*)d_in[3];
  const float* rms_weight = (const float*)d_in[4];
  const float* w_in       = (const float*)d_in[5];
  const float* b_in       = (const float*)d_in[6];
  const float* w_out      = (const float*)d_in[7];
  const float* b_out      = (const float*)d_in[8];
  const float* nu_log     = (const float*)d_in[9];
  const float* theta_log  = (const float*)d_in[10];
  float* out = (float*)d_out;

  bf16* weff    = (bf16*)d_ws;                 // 49*64*64 bf16 = 392 KB
  bf16* win_bf  = weff + TAPS * Cc * Cc;       // 128*64 bf16
  bf16* wout_bf = win_bf + 2 * Cc * Cc;        // 64*64 bf16

  prep_weights<<<(TAPS * Cc * Cc + 3 * Cc * Cc) / 256, 256, 0, stream>>>(
      w_spatial, w_depth, w_in, w_out, weff, win_bf, wout_bf);

  conv_spatial<<<Ll * Hh * (Ww / 32), 256, 0, stream>>>(x, weff, b_depth, out);

  pointwise_in_glu<<<PIXELS / 16, 256, 0, stream>>>(out, win_bf, b_in,
                                                    rms_weight, nu_log);

  lru_scan<<<(Cc * LS) / 256, 256, 0, stream>>>(out, nu_log, theta_log);

  pointwise_out_res<<<PIXELS / 32, 256, 0, stream>>>(out, x, wout_bf, b_out);
}
